// MemoryBasedCollaborativeFiltering_37194416783750
// MI455X (gfx1250) — matrix operand order repaired and checked by reading, not compiled, not run on hardware
//
#include <hip/hip_runtime.h>
#include <math.h>

#define N_USERS 16384
#define N_ITEMS 4096
#define D       64
#define BQ      2048
#define CHUNK   2048
#define NCHUNKS (N_USERS / CHUNK)   // 8
#define NWAVES  8                   // 256 threads / wave32
#define EPSF    1e-8f

typedef __attribute__((ext_vector_type(16))) _Float16 v16h;
typedef __attribute__((ext_vector_type(8)))  _Float16 v8h;
typedef __attribute__((ext_vector_type(8)))  float    v8f;

// ---------------------------------------------------------------------------
// Kernel 1: per-user NaN-masked rating mean, embedding inverse L2 norm, f16
// copy of embeddings. One workgroup per user; streams the 256 MB matrix.
// Writes packed (inv_norm, avg) so kernel 2 fetches both with one b64 load
// and never divides in its inner loop.
// ---------------------------------------------------------------------------
__global__ __launch_bounds__(256) void cf_precompute_kernel(
    const float* __restrict__ ratings,
    const float* __restrict__ emb,
    float2* __restrict__ pna,          // [N_USERS] = (1/norm, avg)
    _Float16* __restrict__ e16) {
  __shared__ float s_sum[256];
  __shared__ float s_cnt[256];
  __shared__ float s_sq[D];

  const int u = blockIdx.x;
  const int t = threadIdx.x;

  const float4* row4 = (const float4*)(ratings + (size_t)u * N_ITEMS);
  float sum = 0.f, cnt = 0.f;
  for (int i = t; i < N_ITEMS / 4; i += 256) {
    float4 v = row4[i];
    if (v.x == v.x) { sum += v.x; cnt += 1.f; }
    if (v.y == v.y) { sum += v.y; cnt += 1.f; }
    if (v.z == v.z) { sum += v.z; cnt += 1.f; }
    if (v.w == v.w) { sum += v.w; cnt += 1.f; }
  }
  s_sum[t] = sum;
  s_cnt[t] = cnt;
  __syncthreads();
  for (int s = 128; s > 0; s >>= 1) {
    if (t < s) { s_sum[t] += s_sum[t + s]; s_cnt[t] += s_cnt[t + s]; }
    __syncthreads();
  }

  if (t < D) {
    float e = emb[(size_t)u * D + t];
    e16[(size_t)u * D + t] = (_Float16)e;
    s_sq[t] = e * e;
  }
  __syncthreads();
  if (t == 0) {
    float ns = 0.f;
    #pragma unroll
    for (int i = 0; i < D; ++i) ns += s_sq[i];
    float c = s_cnt[0];
    float2 p;
    // 1/(mag_u*mag_v + EPS) ~= inv_u*inv_v : EPS=1e-8 is negligible against
    // mag~8; clamp guards a degenerate all-zero embedding.
    p.x = 1.0f / fmaxf(sqrtf(ns), 1e-12f);
    p.y = (c > 0.f) ? (s_sum[0] / c) : 0.f;
    pna[u] = p;
  }
}

// ---------------------------------------------------------------------------
// Kernel 2: fused cosine-sim (WMMA f16->f32) + masked neighbor aggregation.
// grid = (NCHUNKS, BQ/16); block = 256 threads = 8 wave32.
// Per 16x16 tile: two v_wmma_f32_16x16x32_f16 (K=64); 8 scattered rating
// gathers batch-issued before the WMMAs; branchless NaN-safe accumulate with
// NO division/rcp in the loop (inv_u factored out, applied once at the end).
// Unroll x2 so the second tile's loads fill the WMMA->VALU hazard slots.
// ---------------------------------------------------------------------------
__global__ __launch_bounds__(256) void cf_fused_wmma_kernel(
    const float* __restrict__ ratings,
    const _Float16* __restrict__ e16,
    const float2* __restrict__ pna,
    const int* __restrict__ user_idx,
    const int* __restrict__ item_idx,
    float* __restrict__ num_part,
    float* __restrict__ den_part) {
  __shared__ __align__(16) _Float16 s_u[16][D];   // query embedding tile (f16)
  __shared__ int   s_uid[16];
  __shared__ int   s_iid[16];
  __shared__ float s_invu[16];
  __shared__ float s_num[NWAVES][16][16];
  __shared__ float s_den[NWAVES][16][16];

  const int chunk = blockIdx.x;   // neighbor-user chunk
  const int btile = blockIdx.y;   // 16-query tile
  const int t     = threadIdx.x;
  const int wave  = t >> 5;
  const int lane  = t & 31;
  const int half  = lane >> 4;    // 0: lanes 0-15, 1: lanes 16-31
  const int m     = lane & 15;

  if (t < 16) {
    int b = btile * 16 + t;
    int uid = user_idx[b];
    s_uid[t]  = uid;
    s_iid[t]  = item_idx[b];
    s_invu[t] = pna[uid].x;
  }
  __syncthreads();

  // Stage the 16 query rows (128 B each) of the f16 embedding table into LDS.
  if (t < 128) {
    int row = t >> 3, q = t & 7;
    ((uint4*)&s_u[row][0])[q] =
        ((const uint4*)(e16 + (size_t)s_uid[row] * D))[q];
  }
  __syncthreads();

  // Build the A fragments (16x32 f16, contiguous K = 16h+s, same order as B) once; reused for all tiles.
  v16h a_lo, a_hi;
  {
    const _Float16* ur = &s_u[m][0];
    v8h p0 = *(const v8h*)(ur + 16 * half);           // K = 16h .. 16h+7
    v8h p1 = *(const v8h*)(ur + 8 + 16 * half);       // K = 16h+8 .. +7
    v8h p2 = *(const v8h*)(ur + 32 + 16 * half);
    v8h p3 = *(const v8h*)(ur + 40 + 16 * half);
    #pragma unroll
    for (int i = 0; i < 8; ++i) {
      a_lo[i] = p0[i]; a_lo[8 + i] = p1[i];
      a_hi[i] = p2[i]; a_hi[8 + i] = p3[i];
    }
  }

  // Hoist per-row constants out of LDS into registers (loop-invariant).
  int   uid_r[8], ioff_r[8];
  float invu_r[8];
  #pragma unroll
  for (int v = 0; v < 8; ++v) {
    const int row = v + 8 * half;
    uid_r[v]  = s_uid[row];
    ioff_r[v] = s_iid[row];
    invu_r[v] = s_invu[row];
  }

  float accn[8], accd[8];
  #pragma unroll
  for (int v = 0; v < 8; ++v) { accn[v] = 0.f; accd[v] = 0.f; }

  const int n_base = chunk * CHUNK;
  #pragma unroll 2
  for (int tile = wave; tile < CHUNK / 16; tile += NWAVES) {
    const int n = n_base + tile * 16 + m;             // this lane's neighbor

    // B fragment (32x16 f16): column n, contiguous K halves -> plain loads.
    const _Float16* brow = e16 + (size_t)n * D;
    v16h b_lo = *(const v16h*)(brow + 16 * half);       // K  0..31
    v16h b_hi = *(const v16h*)(brow + 32 + 16 * half);  // K 32..63

    // Batch-issue the 8 scattered rating gathers + metadata before the WMMAs
    // so their HBM latency overlaps the matrix pipeline.
    const float* rrow = ratings + (size_t)n * N_ITEMS;
    float r[8];
    #pragma unroll
    for (int v = 0; v < 8; ++v) r[v] = rrow[ioff_r[v]];
    const float2 na = pna[n];                         // (inv_norm_v, avg_r)

    v8f c = {};
    c = __builtin_amdgcn_wmma_f32_16x16x32_f16(false, a_lo, false, b_lo,
                                               (short)0, c, false, false);
    c = __builtin_amdgcn_wmma_f32_16x16x32_f16(false, a_hi, false, b_hi,
                                               (short)0, c, false, false);

    // Branchless, NaN-safe accumulate. sim = c * inv_u * inv_v; the inv_u
    // factor is constant per accumulator -> applied once after the loop.
    #pragma unroll
    for (int v = 0; v < 8; ++v) {
      const bool valid = (r[v] == r[v]) & (n != uid_r[v]);
      const float cs = c[v] * na.x;                   // c * inv_v
      accn[v] += valid ? (r[v] - na.y) * cs : 0.f;
      accd[v] += valid ? fabsf(cs) : 0.f;
    }
  }

  // Per-wave partials -> LDS (unique slot per lane, no atomics); fold in the
  // deferred inv_u scale here.
  #pragma unroll
  for (int v = 0; v < 8; ++v) {
    const int row = v + 8 * half;
    s_num[wave][row][m] = accn[v] * invu_r[v];
    s_den[wave][row][m] = accd[v] * invu_r[v];
  }
  __syncthreads();

  // Fixed-order reduction (deterministic) and per-chunk partial writeback.
  if (t < 16) {
    float nsum = 0.f, dsum = 0.f;
    for (int w = 0; w < NWAVES; ++w)
      for (int mm = 0; mm < 16; ++mm) {
        nsum += s_num[w][t][mm];
        dsum += s_den[w][t][mm];
      }
    int b = btile * 16 + t;
    num_part[(size_t)b * NCHUNKS + chunk] = nsum;
    den_part[(size_t)b * NCHUNKS + chunk] = dsum;
  }
}

// ---------------------------------------------------------------------------
// Kernel 3: reduce chunk partials (fixed order) and finalize prediction.
// ---------------------------------------------------------------------------
__global__ __launch_bounds__(256) void cf_finalize_kernel(
    const float* __restrict__ num_part,
    const float* __restrict__ den_part,
    const float2* __restrict__ pna,
    const int* __restrict__ user_idx,
    float* __restrict__ out) {
  int b = blockIdx.x * blockDim.x + threadIdx.x;
  if (b >= BQ) return;
  float n = 0.f, d = 0.f;
  #pragma unroll
  for (int c = 0; c < NCHUNKS; ++c) {
    n += num_part[(size_t)b * NCHUNKS + c];
    d += den_part[(size_t)b * NCHUNKS + c];
  }
  float au = pna[user_idx[b]].y;
  out[b] = (d == 0.f) ? au : (au + n / d);
}

// ---------------------------------------------------------------------------
extern "C" void kernel_launch(void* const* d_in, const int* in_sizes, int n_in,
                              void* d_out, int out_size, void* d_ws, size_t ws_size,
                              hipStream_t stream) {
  const float* ratings = (const float*)d_in[0];   // [N_USERS, N_ITEMS] f32
  const float* emb     = (const float*)d_in[1];   // [N_USERS, D] f32
  const int*   uidx    = (const int*)d_in[2];     // [BQ]
  const int*   iidx    = (const int*)d_in[3];     // [BQ]
  float* out = (float*)d_out;

  // Workspace layout (all offsets 128B+ aligned):
  char* ws = (char*)d_ws;
  _Float16* e16   = (_Float16*)ws;                              // 2 MB
  float2* pna     = (float2*)(ws + (size_t)N_USERS * D * 2);    // 128 KB
  float* num_part = (float*)(pna + N_USERS);                    // 64 KB
  float* den_part = num_part + (size_t)BQ * NCHUNKS;            // 64 KB

  cf_precompute_kernel<<<N_USERS, 256, 0, stream>>>(ratings, emb, pna, e16);

  dim3 grid(NCHUNKS, BQ / 16);
  cf_fused_wmma_kernel<<<grid, 256, 0, stream>>>(
      ratings, e16, pna, uidx, iidx, num_part, den_part);

  cf_finalize_kernel<<<(BQ + 255) / 256, 256, 0, stream>>>(
      num_part, den_part, pna, uidx, out);
}
